// Qwen3MoE_11854109737682
// MI455X (gfx1250) — matrix operand order repaired and checked by reading, not compiled, not run on hardware
//
#include <hip/hip_runtime.h>
#include <hip/hip_bf16.h>

#define T_TOK 2048
#define H_DIM 2048
#define E_NUM 64
#define I_DIM 768
#define TOPK  8
#define M_TILE 48   // 3 m-subtiles of 16 -> 3-way reuse of each converted B fragment

typedef __attribute__((ext_vector_type(16))) __bf16 bf16x16;
typedef __attribute__((ext_vector_type(8)))  float  f32x8;

// ---------------- WMMA fragment loaders ----------------
// A-matrix 16x32 bf16: lane m = lane&15, kb = 8*(lane>>4);
// elements 0..15 = K[k0+2*kb .. k0+2*kb+15] (contiguous, matches B)
__device__ __forceinline__ bf16x16 load_a_frag(const __bf16* __restrict__ row,
                                               int k0, int kb) {
    union { bf16x16 v; uint4 q[2]; } u;
    u.q[0] = *reinterpret_cast<const uint4*>(row + k0 + 2 * kb);
    u.q[1] = *reinterpret_cast<const uint4*>(row + k0 + 2 * kb + 8);
    return u.v;
}

// B-matrix 32x16 bf16: lane n = lane&15; K = k0 + 16*(lane>>4) + [0..15]
// (contiguous along K in memory; converted fp32 -> bf16 on the fly)
__device__ __forceinline__ bf16x16 load_b_frag(const float* __restrict__ p) {
    const float4* q = reinterpret_cast<const float4*>(p);
    float4 a = q[0], b = q[1], c = q[2], d = q[3];
    bf16x16 r;
    r[0]  = (__bf16)a.x; r[1]  = (__bf16)a.y; r[2]  = (__bf16)a.z; r[3]  = (__bf16)a.w;
    r[4]  = (__bf16)b.x; r[5]  = (__bf16)b.y; r[6]  = (__bf16)b.z; r[7]  = (__bf16)b.w;
    r[8]  = (__bf16)c.x; r[9]  = (__bf16)c.y; r[10] = (__bf16)c.z; r[11] = (__bf16)c.w;
    r[12] = (__bf16)d.x; r[13] = (__bf16)d.y; r[14] = (__bf16)d.z; r[15] = (__bf16)d.w;
    return r;
}

__device__ __forceinline__ f32x8 zero8() {
    f32x8 z = {0.f, 0.f, 0.f, 0.f, 0.f, 0.f, 0.f, 0.f};
    return z;
}

#define WMMA_BF16(A, B, C) \
    __builtin_amdgcn_wmma_f32_16x16x32_bf16(false, (A), false, (B), (short)0, (C), false, false)

// ---------------- Kernel 0: X fp32 -> bf16 (once, 8 MB) ----------------
__global__ void cvt_x_bf16(const float* __restrict__ X, __bf16* __restrict__ Xb, int n) {
    int i = (blockIdx.x * 256 + threadIdx.x) * 8;
    if (i >= n) return;
    float4 a = *reinterpret_cast<const float4*>(X + i);
    float4 b = *reinterpret_cast<const float4*>(X + i + 4);
    union { __bf16 h[8]; uint4 q; } u;
    u.h[0] = (__bf16)a.x; u.h[1] = (__bf16)a.y; u.h[2] = (__bf16)a.z; u.h[3] = (__bf16)a.w;
    u.h[4] = (__bf16)b.x; u.h[5] = (__bf16)b.y; u.h[6] = (__bf16)b.z; u.h[7] = (__bf16)b.w;
    *reinterpret_cast<uint4*>(Xb + i) = u.q;
}

// ---------------- Kernel 1: gate GEMV + top-k + softmax ----------------
__global__ void moe_route(const float* __restrict__ X,
                          const float* __restrict__ G,
                          int* __restrict__ sel_e,
                          float* __restrict__ sel_w) {
    const int t = blockIdx.x;       // token
    const int e = threadIdx.x;      // expert (blockDim = 64)
    __shared__ float xr[H_DIM];
    __shared__ float lg[E_NUM];

    for (int c = e; c < H_DIM; c += 64)
        xr[c] = X[(size_t)t * H_DIM + c];
    __syncthreads();

    const float* g = G + (size_t)e * H_DIM;
    float acc = 0.f;
    for (int c = 0; c < H_DIM; c += 4) {
        float4 gv = *reinterpret_cast<const float4*>(g + c);
        acc += xr[c] * gv.x + xr[c + 1] * gv.y + xr[c + 2] * gv.z + xr[c + 3] * gv.w;
    }
    lg[e] = acc;
    __syncthreads();

    if (e == 0) {
        float v[TOPK]; int ix[TOPK];
        for (int k = 0; k < TOPK; ++k) {
            float best = -3.4e38f; int bi = 0;
            for (int j = 0; j < E_NUM; ++j)
                if (lg[j] > best) { best = lg[j]; bi = j; }
            v[k] = best; ix[k] = bi; lg[bi] = -3.4e38f;
        }
        float m0 = v[0], s = 0.f, p[TOPK];
        for (int k = 0; k < TOPK; ++k) { p[k] = __expf(v[k] - m0); s += p[k]; }
        float inv = 1.f / s;
        for (int k = 0; k < TOPK; ++k) {
            sel_e[t * TOPK + k] = ix[k];
            sel_w[t * TOPK + k] = p[k] * inv;
        }
    }
}

// ---------------- Kernel 2: deterministic per-expert bucketing ----------------
__global__ void moe_bucket(const int* __restrict__ sel_e,
                           const float* __restrict__ sel_w,
                           int* __restrict__ counts,
                           int* __restrict__ tok_id,
                           float* __restrict__ tok_w) {
    const int e = blockIdx.x;       // one wave (32 lanes) per expert
    const int lane = threadIdx.x;
    int count = 0;
    for (int base = 0; base < T_TOK; base += 32) {
        int t = base + lane;
        float w = 0.f; bool has = false;
#pragma unroll
        for (int k = 0; k < TOPK; ++k) {
            if (sel_e[t * TOPK + k] == e) { has = true; w = sel_w[t * TOPK + k]; }
        }
        unsigned mask = __builtin_amdgcn_ballot_w32(has);
        int pre = __popc(mask & ((1u << lane) - 1u));
        if (has) {
            tok_id[e * T_TOK + count + pre] = t;
            tok_w [e * T_TOK + count + pre] = w;
        }
        count += __popc(mask);
    }
    if (lane == 0) counts[e] = count;
}

// ---------------- Kernel 3: zero output ----------------
__global__ void zero_out(float* __restrict__ o, int n) {
    int i = (blockIdx.x * 256 + threadIdx.x) * 4;
    if (i < n) *reinterpret_cast<float4*>(o + i) = make_float4(0.f, 0.f, 0.f, 0.f);
}

// ---------------- Kernel 4: expert SwiGLU MLP with WMMA ----------------
__global__ __launch_bounds__(256) void moe_expert(
        const __bf16* __restrict__ Xb,      // [T,H] bf16 (pre-converted)
        const float* __restrict__ w1,
        const float* __restrict__ w3,
        const float* __restrict__ w2,
        const int* __restrict__ counts,
        const int* __restrict__ tok_id,
        const float* __restrict__ tok_w,
        float* __restrict__ out) {
    const int e    = blockIdx.y;
    const int tile = blockIdx.x;
    const int cnt  = counts[e];
    const int row0 = tile * M_TILE;
    if (row0 >= cnt) return;   // block-uniform exit (EXEC stays all-ones for WMMA)

    extern __shared__ char smem[];
    __bf16* Xl = reinterpret_cast<__bf16*>(smem);                            // 48 x 2048 bf16
    __bf16* Hl = reinterpret_cast<__bf16*>(smem + M_TILE * H_DIM * 2);       // 48 x 768 bf16
    int*    Tk = reinterpret_cast<int*>  (smem + M_TILE * H_DIM * 2 + M_TILE * I_DIM * 2);
    float*  Wt = reinterpret_cast<float*>(Tk + M_TILE);

    const int tid  = threadIdx.x;
    const int lane = tid & 31;
    const int wv   = tid >> 5;     // wave id 0..7
    const int hi   = lane >> 4;    // lane-half
    const int nn   = lane & 15;
    const int kb   = hi * 8;       // A-frag K sub-offset

    if (tid < M_TILE) {
        int idx = row0 + tid;
        if (idx < cnt) { Tk[tid] = tok_id[e * T_TOK + idx]; Wt[tid] = tok_w[e * T_TOK + idx]; }
        else           { Tk[tid] = tok_id[e * T_TOK + row0]; Wt[tid] = 0.f; }
    }
    __syncthreads();

    // ---- Stage gathered bf16 token tile into LDS via async-to-LDS DMA ----
    // Each pass: 256 threads x 16 B = one 2048-elem bf16 row (4 KB).
    {
        unsigned lbase = (unsigned)(uintptr_t)Xl + (unsigned)(tid * 16);
        for (int r = 0; r < M_TILE; ++r) {
            const char* src = reinterpret_cast<const char*>(Xb + (size_t)Tk[r] * H_DIM)
                              + (size_t)tid * 16;
            unsigned laddr = lbase + (unsigned)(r * (H_DIM * 2));
            asm volatile("global_load_async_to_lds_b128 %0, %1, off"
                         :: "v"(laddr), "v"(src) : "memory");
        }
        asm volatile("s_wait_asynccnt 0x0" ::: "memory");
    }
    __syncthreads();

    // ---- Phase 1: h = silu(X w1^T) * (X w3^T), tile [48 x I] -> LDS bf16 ----
    const __bf16* arow0 = Xl + (size_t)(nn)      * H_DIM;   // m-subtile 0 (rows 0..15)
    const __bf16* arow1 = Xl + (size_t)(16 + nn) * H_DIM;   // m-subtile 1 (rows 16..31)
    const __bf16* arow2 = Xl + (size_t)(32 + nn) * H_DIM;   // m-subtile 2 (rows 32..47)
    for (int it = wv; it < I_DIM / 16; it += 8) {
        const float* b1row = w1 + ((size_t)e * I_DIM + it * 16 + nn) * H_DIM;
        const float* b3row = w3 + ((size_t)e * I_DIM + it * 16 + nn) * H_DIM;
        f32x8 g0 = zero8(), g1 = zero8(), g2 = zero8();
        f32x8 u0 = zero8(), u1 = zero8(), u2 = zero8();
#pragma unroll 2
        for (int k0 = 0; k0 < H_DIM; k0 += 32) {
            bf16x16 a0 = load_a_frag(arow0, k0, kb);
            bf16x16 a1 = load_a_frag(arow1, k0, kb);
            bf16x16 a2 = load_a_frag(arow2, k0, kb);
            bf16x16 b1 = load_b_frag(b1row + k0 + 16 * hi);
            g0 = WMMA_BF16(a0, b1, g0);
            g1 = WMMA_BF16(a1, b1, g1);          // 3-way reuse of converted B fragment
            g2 = WMMA_BF16(a2, b1, g2);
            bf16x16 b3 = load_b_frag(b3row + k0 + 16 * hi);
            u0 = WMMA_BF16(a0, b3, u0);
            u1 = WMMA_BF16(a1, b3, u1);
            u2 = WMMA_BF16(a2, b3, u2);
        }
#pragma unroll
        for (int r = 0; r < 8; ++r) {   // C layout: m = r + 8*hi, n = lane&15
            float gv0 = g0[r], uv0 = u0[r];
            float gv1 = g1[r], uv1 = u1[r];
            float gv2 = g2[r], uv2 = u2[r];
            float hv0 = (gv0 / (1.f + __expf(-gv0))) * uv0;
            float hv1 = (gv1 / (1.f + __expf(-gv1))) * uv1;
            float hv2 = (gv2 / (1.f + __expf(-gv2))) * uv2;
            Hl[(r + 8 * hi) * I_DIM + it * 16 + nn]      = (__bf16)hv0;
            Hl[(16 + r + 8 * hi) * I_DIM + it * 16 + nn] = (__bf16)hv1;
            Hl[(32 + r + 8 * hi) * I_DIM + it * 16 + nn] = (__bf16)hv2;
        }
    }
    __syncthreads();

    // ---- Phase 2: out[tok] += (h w2^T) * route_weight ----
    float wl0[8], wl1[8], wl2[8];
    int   tk0[8], tk1[8], tk2[8];
#pragma unroll
    for (int r = 0; r < 8; ++r) {
        wl0[r] = Wt[r + 8 * hi];       tk0[r] = Tk[r + 8 * hi];
        wl1[r] = Wt[16 + r + 8 * hi];  tk1[r] = Tk[16 + r + 8 * hi];
        wl2[r] = Wt[32 + r + 8 * hi];  tk2[r] = Tk[32 + r + 8 * hi];
    }

    const __bf16* hrow0 = Hl + (size_t)(nn)      * I_DIM;
    const __bf16* hrow1 = Hl + (size_t)(16 + nn) * I_DIM;
    const __bf16* hrow2 = Hl + (size_t)(32 + nn) * I_DIM;
    for (int ht = wv; ht < H_DIM / 16; ht += 8) {
        const int hcol = ht * 16 + nn;
        const float* b2row = w2 + ((size_t)e * H_DIM + hcol) * I_DIM;
        f32x8 acc0 = zero8(), acc1 = zero8(), acc2 = zero8();
#pragma unroll 2
        for (int k0 = 0; k0 < I_DIM; k0 += 32) {
            bf16x16 a0 = load_a_frag(hrow0, k0, kb);
            bf16x16 a1 = load_a_frag(hrow1, k0, kb);
            bf16x16 a2 = load_a_frag(hrow2, k0, kb);
            bf16x16 b  = load_b_frag(b2row + k0 + 16 * hi);
            acc0 = WMMA_BF16(a0, b, acc0);
            acc1 = WMMA_BF16(a1, b, acc1);       // 3-way reuse of converted B fragment
            acc2 = WMMA_BF16(a2, b, acc2);
        }
#pragma unroll
        for (int r = 0; r < 8; ++r) {
            atomicAdd(out + (size_t)tk0[r] * H_DIM + hcol, acc0[r] * wl0[r]);
            atomicAdd(out + (size_t)tk1[r] * H_DIM + hcol, acc1[r] * wl1[r]);
            atomicAdd(out + (size_t)tk2[r] * H_DIM + hcol, acc2[r] * wl2[r]);
        }
    }
}

// ---------------- Host launcher ----------------
extern "C" void kernel_launch(void* const* d_in, const int* in_sizes, int n_in,
                              void* d_out, int out_size, void* d_ws, size_t ws_size,
                              hipStream_t stream) {
    const float* X  = (const float*)d_in[0];   // [T,H]
    const float* G  = (const float*)d_in[1];   // [E,H]
    const float* w1 = (const float*)d_in[2];   // [E,I,H]
    const float* w3 = (const float*)d_in[3];   // [E,I,H]
    const float* w2 = (const float*)d_in[4];   // [E,H,I]
    float* out = (float*)d_out;                // [T,H]

    char* ws = (char*)d_ws;
    size_t off = 0;
    int*   sel_e  = (int*)  (ws + off); off += (size_t)T_TOK * TOPK * 4;
    float* sel_w  = (float*)(ws + off); off += (size_t)T_TOK * TOPK * 4;
    int*   counts = (int*)  (ws + off); off += 256;
    int*   tok_id = (int*)  (ws + off); off += (size_t)E_NUM * T_TOK * 4;
    float* tok_w  = (float*)(ws + off); off += (size_t)E_NUM * T_TOK * 4;
    __bf16* Xb    = (__bf16*)(ws + off); off += (size_t)T_TOK * H_DIM * 2;

    int n = T_TOK * H_DIM;
    cvt_x_bf16<<<(n / 8 + 255) / 256, 256, 0, stream>>>(X, Xb, n);
    moe_route <<<T_TOK, 64, 0, stream>>>(X, G, sel_e, sel_w);
    moe_bucket<<<E_NUM, 32, 0, stream>>>(sel_e, sel_w, counts, tok_id, tok_w);
    zero_out<<<(n / 4 + 255) / 256, 256, 0, stream>>>(out, n);

    dim3 grid((T_TOK + M_TILE - 1) / M_TILE, E_NUM);   // tiles x experts
    size_t smem = (size_t)M_TILE * H_DIM * 2 + (size_t)M_TILE * I_DIM * 2
                + M_TILE * 4 + M_TILE * 4;
    moe_expert<<<grid, 256, smem, stream>>>(Xb, w1, w3, w2, counts, tok_id, tok_w, out);
}
